// FlowAE_20486994002690
// MI455X (gfx1250) — matrix operand order repaired and checked by reading, not compiled, not run on hardware
//
#include <hip/hip_runtime.h>
#include <hip/hip_bf16.h>

// ---------------------------------------------------------------------------
// FlowAE on gfx1250 (MI455X, wave32, WMMA + TDM).
// Shapes: B=16, T=2048, D_IN=16, D_MODEL=256, D_FF=1024, N = B*T = 32768.
// Big GEMMs (N x 256 x 1024, N x 1024 x 256) run on v_wmma_f32_16x16x32_f16
// with f32 accumulation. 32 tokens per WG: every weight (B) fragment feeds
// two WMMAs and weight L2 traffic is halved vs a 16-token tile.
// ---------------------------------------------------------------------------

typedef __attribute__((ext_vector_type(16))) _Float16 v16h;
typedef __attribute__((ext_vector_type(8)))  _Float16 v8h;
typedef __attribute__((ext_vector_type(8)))  float    v8f;
typedef __attribute__((ext_vector_type(4)))  unsigned int u32x4;
typedef __attribute__((ext_vector_type(8)))  int i32x8;
typedef __attribute__((ext_vector_type(4)))  int i32x4;

#define NB 16
#define NT 2048
#define NTOK (NB * NT)          // 32768
#define DMODEL 256
#define DFF 1024
#define TPW 32                  // tokens per workgroup (two M-tiles)
#define H0S (DMODEL + 8)        // padded LDS strides (bank spread)
#define H1S (DFF + 8)
#define H2S (DMODEL + 4)

#if __has_builtin(__builtin_amdgcn_tensor_load_to_lds) && \
    __has_builtin(__builtin_amdgcn_s_wait_tensorcnt)
#define HAVE_TDM 1
#else
#define HAVE_TDM 0
#endif

#if HAVE_TDM
// 1-D contiguous TDM copy: nelem elements of 4 bytes, global -> LDS.
// D# per CDNA5 ISA ch.8: group0 {count=1 | lds_addr | global_addr | type=2},
// group1 {data_size=4B, tensor_dim0=nelem, tensor_dim1=1, tile_dim0=nelem,
//         tile_dim1=1, dim0_stride=nelem}. Groups 2/3 zero (<=2D tensor).
static __device__ inline void tdm_copy_1d_b32(unsigned lds_byte_addr,
                                              const void* gptr, unsigned nelem)
{
    const unsigned long long g = (unsigned long long)gptr;
    u32x4 g0;
    g0[0] = 1u;                                           // count=1, user desc
    g0[1] = lds_byte_addr;                                // lds_addr
    g0[2] = (unsigned)(g & 0xffffffffu);                  // global_addr[31:0]
    g0[3] = (unsigned)((g >> 32) & 0x01ffffffu) | (2u << 30); // addr[56:32]|type=2
    i32x8 g1;
    g1[0] = (int)(2u << 16);                              // data_size: 4 bytes
    g1[1] = (int)((nelem & 0xffffu) << 16);               // tensor_dim0[15:0]
    g1[2] = (int)((nelem >> 16) | (1u << 16));            // dim0[31:16] | dim1=1
    g1[3] = (int)(nelem << 16);                           // tile_dim0 = nelem
    g1[4] = 1;                                            // tile_dim1 = 1
    g1[5] = (int)nelem;                                   // dim0_stride
    g1[6] = 0;
    g1[7] = 0;
    i32x4 z4 = {};
#if __clang_major__ >= 23
    i32x8 z8 = {};
    __builtin_amdgcn_tensor_load_to_lds(g0, g1, z4, z4, z8, 0);
#else
    __builtin_amdgcn_tensor_load_to_lds(g0, g1, z4, z4, 0);
#endif
}
#endif

// ---------------------------------------------------------------------------
// prep: W1[256][1024] -> W1t[n][k] f16 ; W2[1024][256] -> W2t[n][k] f16.
// n-major layout makes each WMMA B-fragment one contiguous 32B load per lane.
// ---------------------------------------------------------------------------
__global__ __launch_bounds__(256) void prep_kernel(
    const float* __restrict__ W1, const float* __restrict__ W2,
    _Float16* __restrict__ W1t, _Float16* __restrict__ W2t)
{
    int i = blockIdx.x * 256 + threadIdx.x;          // grid covers 2*262144
    if (i < DMODEL * DFF) {
        int k = i >> 10, n = i & (DFF - 1);          // W1[k][n]
        W1t[n * DMODEL + k] = (_Float16)W1[i];
    } else {
        int j = i - DMODEL * DFF;
        int k = j >> 8, n = j & (DMODEL - 1);        // W2[k][n]
        W2t[n * DFF + k] = (_Float16)W2[j];
    }
}

// Build a 16x32 f16 A-fragment from row-major LDS (stride in halfs).
// Contiguous k order (same in B loads): lane l holds row m=l%16; elements 0..15 = K kb..kb+15
// where kb = k0 + (l>>4)*16.
template <int STRIDE>
static __device__ inline v16h load_A(const _Float16* lds, int lane, int k0)
{
    int m  = lane & 15;
    int kb = k0 + ((lane >> 4) << 4);
    const v8h lo = *(const v8h*)(lds + m * STRIDE + kb);
    const v8h hi = *(const v8h*)(lds + m * STRIDE + kb + 8);
    v16h a;
#pragma unroll
    for (int i = 0; i < 8; ++i) { a[i] = lo[i]; a[8 + i] = hi[i]; }
    return a;
}

static __device__ inline v8f wmma_f16(const v16h a, const v16h b, const v8f c)
{
    return __builtin_amdgcn_wmma_f32_16x16x32_f16(false, a, false, b,
                                                  (short)0, c, false, false);
}

// ---------------------------------------------------------------------------
// mlp: per-WG = 32 tokens (2 M-tiles), 256 threads (8 waves).
//   TDM stages input tile + W_skew into LDS (async, TENSORcnt).
//   embed (scalar f32, K=16)  -> h0 f16 LDS
//   GEMM2 relu(h0 @ W1 + b1)  -> h1 f16 LDS   (wave owns 128 cols: each B
//                                              fragment feeds 2 WMMAs)
//   GEMM3 h1 @ W2 + b2        -> h2 f32 LDS   (wave owns 32 cols: 2x2 tile)
//   skew head -> derived flow params; per-WG pool partial (deterministic).
// h0 (dead after GEMM2) is unioned with h2 to keep LDS ~106KB.
// ---------------------------------------------------------------------------
__global__ __launch_bounds__(256) void mlp_kernel(
    const float* __restrict__ inputs,
    const float* __restrict__ W_embed, const float* __restrict__ b_embed,
    const _Float16* __restrict__ W1t, const float* __restrict__ b1,
    const _Float16* __restrict__ W2t, const float* __restrict__ b2,
    const float* __restrict__ W_skew, const float* __restrict__ b_skew,
    float* __restrict__ params, float* __restrict__ pool_partial)
{
    __shared__ float    in_lds[TPW][16];
    __shared__ float    wsk_lds[DMODEL * 4];
    __shared__ union {
        _Float16 h0[TPW][H0S];      // 16.9 KB, dead after GEMM2
        float    h2[TPW][H2S];      // 33.3 KB
    } u;
    __shared__ _Float16 h1s[TPW][H1S];   // 66 KB
    __shared__ float    sp_lds[TPW][4];

    const int tid  = threadIdx.x;
    const int wave = tid >> 5;
    const int lane = tid & 31;
    const int tok0 = blockIdx.x * TPW;

    // ---- stage input tile [32][16] (contiguous 2KB) and W_skew (4KB)
#if HAVE_TDM
    if (wave == 0) {
        // generic->LDS address: low 32 bits of the generic address are the
        // LDS byte offset (flat-aperture mapping, ISA 10.2).
        tdm_copy_1d_b32((unsigned)(unsigned long long)(size_t)&in_lds[0][0],
                        inputs + (size_t)tok0 * 16, TPW * 16);
        tdm_copy_1d_b32((unsigned)(unsigned long long)(size_t)&wsk_lds[0],
                        W_skew, DMODEL * 4);
        __builtin_amdgcn_s_wait_tensorcnt(0);
    }
#else
    {
        int m = tid >> 4, k = tid & 15;
        in_lds[m][k] = inputs[(tok0 + m) * 16 + k];
        in_lds[16 + m][k] = inputs[(tok0 + 16 + m) * 16 + k];
        wsk_lds[tid] = W_skew[tid];
        wsk_lds[256 + tid] = W_skew[256 + tid];
        wsk_lds[512 + tid] = W_skew[512 + tid];
        wsk_lds[768 + tid] = W_skew[768 + tid];
    }
#endif
    __syncthreads();

    // ---- embed: h0 = inputs @ W_embed + b_embed (K=16, ~1/64 of the flops)
    {
        const int n = tid;                // 0..255, coalesced W_embed reads
#pragma unroll 1
        for (int mh = 0; mh < TPW; mh += 16) {
            float acc[16];
#pragma unroll
            for (int m = 0; m < 16; ++m) acc[m] = b_embed[n];
#pragma unroll
            for (int k = 0; k < 16; ++k) {
                float w = W_embed[k * DMODEL + n];
#pragma unroll
                for (int m = 0; m < 16; ++m)
                    acc[m] = fmaf(in_lds[mh + m][k], w, acc[m]);
            }
#pragma unroll
            for (int m = 0; m < 16; ++m) u.h0[mh + m][n] = (_Float16)acc[m];
        }
    }
    __syncthreads();

    // ---- GEMM2: h1 = relu(h0[32,256] @ W1[256,1024] + b1); wave w -> cols
    //      [w*128, w*128+128). Each B fragment feeds both M-tiles.
    {
        const int ncol0 = wave * 128;
        const int bn_l  = lane & 15;
        v8f acc0[8] = {}, acc1[8] = {};
#pragma unroll 1
        for (int kk = 0; kk < DMODEL / 32; ++kk) {
            const int k0 = kk * 32;
            const int bk = k0 + ((lane >> 4) << 4);
            const v16h a0 = load_A<H0S>(&u.h0[0][0],  lane, k0);
            const v16h a1 = load_A<H0S>(&u.h0[16][0], lane, k0);
#pragma unroll
            for (int j = 0; j < 8; ++j) {
                const v16h b = *(const v16h*)(W1t + (ncol0 + j * 16 + bn_l) * DMODEL + bk);
                acc0[j] = wmma_f16(a0, b, acc0[j]);
                acc1[j] = wmma_f16(a1, b, acc1[j]);
            }
        }
        // bias + relu -> f16 LDS (both M-tiles)
#pragma unroll
        for (int j = 0; j < 8; ++j) {
            const int n  = ncol0 + j * 16 + (lane & 15);
            const float bb = b1[n];
            const int mh = (lane >> 4) << 3;
#pragma unroll
            for (int v = 0; v < 8; ++v) {
                float x0 = acc0[j][v] + bb;
                float x1 = acc1[j][v] + bb;
                h1s[mh + v][n]      = (_Float16)(x0 > 0.f ? x0 : 0.f);
                h1s[16 + mh + v][n] = (_Float16)(x1 > 0.f ? x1 : 0.f);
            }
        }
    }
    __syncthreads();     // h1 visible; u.h0 dead -> u.h2 reusable after this

    // ---- GEMM3: h2 = h1[32,1024] @ W2[1024,256] + b2; wave w -> cols
    //      [w*32, w*32+32). 2x2 register tile: A and B each reused twice.
    {
        const int ncol0 = wave * 32;
        const int bn_l  = lane & 15;
        v8f c00 = {}, c01 = {}, c10 = {}, c11 = {};
#pragma unroll 2
        for (int kk = 0; kk < DFF / 32; ++kk) {
            const int k0 = kk * 32;
            const int bk = k0 + ((lane >> 4) << 4);
            const v16h b0 = *(const v16h*)(W2t + (ncol0 + bn_l) * DFF + bk);
            const v16h b1v = *(const v16h*)(W2t + (ncol0 + 16 + bn_l) * DFF + bk);
            const v16h a0 = load_A<H1S>(&h1s[0][0],  lane, k0);
            const v16h a1 = load_A<H1S>(&h1s[16][0], lane, k0);
            c00 = wmma_f16(a0, b0, c00);
            c01 = wmma_f16(a0, b1v, c01);
            c10 = wmma_f16(a1, b0, c10);
            c11 = wmma_f16(a1, b1v, c11);
        }
        const int mh = (lane >> 4) << 3;
        const int n0 = ncol0 + bn_l;
        const int n1 = ncol0 + 16 + bn_l;
        const float bb0 = b2[n0], bb1 = b2[n1];
#pragma unroll
        for (int v = 0; v < 8; ++v) {
            u.h2[mh + v][n0]      = c00[v] + bb0;
            u.h2[mh + v][n1]      = c01[v] + bb1;
            u.h2[16 + mh + v][n0] = c10[v] + bb0;
            u.h2[16 + mh + v][n1] = c11[v] + bb1;
        }
    }
    __syncthreads();

    // ---- skew head: sp[32][4] = h2 @ W_skew + b_skew (W_skew from LDS)
    if (tid < 128) {
        const int m = tid >> 2, c = tid & 3;
        float s = b_skew[c];
        for (int k = 0; k < DMODEL; ++k) s = fmaf(u.h2[m][k], wsk_lds[k * 4 + c], s);
        sp_lds[m][c] = s;
    }
    // ---- per-WG pool partial (all 32 tokens of this WG are in one batch)
    {
        const int n = tid;
        float s = 0.f;
#pragma unroll
        for (int m = 0; m < TPW; ++m) s += u.h2[m][n];
        pool_partial[blockIdx.x * DMODEL + n] = s;
    }
    __syncthreads();

    // ---- derived flow params: (xi, 1/om, al+eps, 1/(al+eps), rho*norm)
    if (tid < TPW) {
        const int g = tok0 + tid;
        const float xi  = expf(sp_lds[tid][0]);
        const float om  = expf(sp_lds[tid][1]);
        const float al  = expf(sp_lds[tid][2]);
        const float rho = expf(sp_lds[tid][3]);
        const float alp = al + 1e-12f;
        const float inv_alp = 1.0f / alp;
        const float coef = rho / (om * (al + inv_alp));   // rho * norm
        float* P = params + (size_t)g * 8;
        P[0] = xi; P[1] = 1.0f / om; P[2] = alp; P[3] = inv_alp; P[4] = coef;
        P[5] = 0.f; P[6] = 0.f; P[7] = 0.f;
    }
}

// ---------------------------------------------------------------------------
// bf: Bf[b] = exp(mean_t(h2) @ W_bf + b_bf), reducing NT/TPW WG partials/batch.
// ---------------------------------------------------------------------------
__global__ __launch_bounds__(256) void bf_kernel(
    const float* __restrict__ pool_partial,
    const float* __restrict__ W_bf, const float* __restrict__ b_bf,
    float* __restrict__ Bf)
{
    __shared__ float red[256];
    const int b = blockIdx.x, n = threadIdx.x;
    float s = 0.f;
    for (int j = 0; j < NT / TPW; ++j)                  // 64 partials
        s += pool_partial[(b * (NT / TPW) + j) * DMODEL + n];
    red[n] = (s * (1.0f / (float)NT)) * W_bf[n];
    __syncthreads();
    for (int off = 128; off > 0; off >>= 1) {
        if (n < off) red[n] += red[n + off];
        __syncthreads();
    }
    if (n == 0) Bf[b] = __expf(red[0] + b_bf[0]);
}

// ---------------------------------------------------------------------------
// flow: flow_pred[b,t] = Bf[b] + sum_{i<=t} p[b,i] * K(b,i,t).
// One thread per output; a block covers 256 consecutive t of one batch, so
// the per-i parameters are staged into LDS in 256-entry chunks and read back
// as conflict-free broadcasts. Inner loop: fmas + one __expf, no divides.
// ---------------------------------------------------------------------------
__global__ __launch_bounds__(256) void flow_kernel(
    const float* __restrict__ precip, const float* __restrict__ flow_true,
    const float* __restrict__ params, const float* __restrict__ Bf,
    float* __restrict__ out, float* __restrict__ mse_partial)
{
    __shared__ float sxi[256], sinvom[256], salp[256], sinvalp[256];
    __shared__ float scoef[256], spr[256];
    __shared__ float red[256];

    const int tid = threadIdx.x;
    const int gid = blockIdx.x * 256 + tid;             // 0..32767
    const int b = gid >> 11;
    const int t = gid & (NT - 1);
    const int t0 = (blockIdx.x * 256) & (NT - 1);       // block-uniform
    const float* P = params + ((size_t)b << 11) * 8;
    const float* p = precip + ((size_t)b << 11);

    float acc = Bf[b];
    const int nchunk = (t0 >> 8) + 1;                   // chunks cover [0, t0+255]
    for (int c = 0; c < nchunk; ++c) {
        const int i0 = c << 8;
        __syncthreads();
        {
            const int i = i0 + tid;                     // always < NT
            const float4 q = *(const float4*)(P + (size_t)i * 8);
            sxi[tid] = q.x; sinvom[tid] = q.y; salp[tid] = q.z; sinvalp[tid] = q.w;
            scoef[tid] = P[(size_t)i * 8 + 4];
            spr[tid] = p[i];
        }
        __syncthreads();
        const int iend = t - i0;                        // local inclusive end
        const int ilim = iend > 255 ? 255 : iend;       // may be <0 -> skip
        for (int il = 0; il <= ilim; ++il) {
            const float d = (float)(t - (i0 + il)) - sxi[il];
            const float s = (d > 0.f) ? 1.f : ((d < 0.f) ? -1.f : 0.f);
            const float f = (d > 0.f) ? salp[il] : ((d < 0.f) ? sinvalp[il] : 1.f);
            const float K = scoef[il] * __expf(-(d * sinvom[il]) * s * f);
            acc = fmaf(spr[il], K, acc);
        }
    }
    out[2 + gid] = acc;

    const float e = acc - flow_true[gid];
    red[tid] = e * e;
    __syncthreads();
    for (int off = 128; off > 0; off >>= 1) {
        if (tid < off) red[tid] += red[tid + off];
        __syncthreads();
    }
    if (tid == 0) mse_partial[blockIdx.x] = red[0];
}

__global__ void finalize_kernel(const float* __restrict__ mse_partial,
                                float* __restrict__ out)
{
    float s = 0.f;
    for (int i = 0; i < NTOK / 256; ++i) s += mse_partial[i];
    const float mse = s * (1.0f / (float)NTOK);
    out[0] = mse;        // loss
    out[1] = -mse;       // log_lik
}

// ---------------------------------------------------------------------------
extern "C" void kernel_launch(void* const* d_in, const int* in_sizes, int n_in,
                              void* d_out, int out_size, void* d_ws, size_t ws_size,
                              hipStream_t stream)
{
    const float* inputs    = (const float*)d_in[0];
    const float* precip    = (const float*)d_in[1];
    const float* flow_true = (const float*)d_in[2];
    const float* W_embed   = (const float*)d_in[3];
    const float* b_embed   = (const float*)d_in[4];
    const float* W1        = (const float*)d_in[5];
    const float* b1        = (const float*)d_in[6];
    const float* W2        = (const float*)d_in[7];
    const float* b2        = (const float*)d_in[8];
    const float* W_skew    = (const float*)d_in[9];
    const float* b_skew    = (const float*)d_in[10];
    const float* W_bf      = (const float*)d_in[11];
    const float* b_bf      = (const float*)d_in[12];

    char* ws = (char*)d_ws;
    _Float16* W1t        = (_Float16*)(ws);                       // 512 KB
    _Float16* W2t        = (_Float16*)(ws + (512u << 10));        // 512 KB
    float* params        = (float*)(ws + (1024u << 10));          // 1 MB
    float* pool_partial  = (float*)(ws + (2048u << 10));          // 1 MB
    float* Bf            = (float*)(ws + (4096u << 10));          // 64 B
    float* mse_partial   = (float*)(ws + (4096u << 10) + 256);    // 512 B

    float* out = (float*)d_out;

    prep_kernel<<<(2 * DMODEL * DFF) / 256, 256, 0, stream>>>(W1, W2, W1t, W2t);
    mlp_kernel<<<NTOK / TPW, 256, 0, stream>>>(
        inputs, W_embed, b_embed, W1t, b1, W2t, b2, W_skew, b_skew,
        params, pool_partial);
    bf_kernel<<<NB, 256, 0, stream>>>(pool_partial, W_bf, b_bf, Bf);
    flow_kernel<<<NTOK / 256, 256, 0, stream>>>(
        precip, flow_true, params, Bf, out, mse_partial);
    finalize_kernel<<<1, 1, 0, stream>>>(mse_partial, out);
}
